// GraphAttentionNetwork_5720896438796
// MI455X (gfx1250) — matrix operand order repaired and checked by reading, not compiled, not run on hardware
//
#include <hip/hip_runtime.h>
#include <hip/hip_bf16.h>
#include <cmath>

// Problem constants (from reference)
constexpr int B = 16, N = 256, E = 4096, D = 1024, H = 8, F = 128;

typedef __attribute__((ext_vector_type(16))) _Float16 v16h;
typedef __attribute__((ext_vector_type(8)))  _Float16 v8h;
typedef __attribute__((ext_vector_type(2)))  _Float16 hh2;
typedef __attribute__((ext_vector_type(8)))  float    v8f;
typedef __attribute__((ext_vector_type(2)))  float    ff2;

__device__ __forceinline__ float leaky(float x) { return (x >= 0.f) ? x : 0.1f * x; }

// ---------------------------------------------------------------------------
// K_proj: fold w_attn into Wg / We columns (deletes the 137 GF e_proj GEMM):
//   wgw1[d,h] = sum_f Wg[d, h*F+f] * w_attn[f]         (same for w2, We/w3)
// ---------------------------------------------------------------------------
__global__ void k_proj(const float* __restrict__ Wg, const float* __restrict__ We,
                       const float* __restrict__ w_attn,
                       float* __restrict__ wgw1, float* __restrict__ wgw2,
                       float* __restrict__ wew3) {
  int idx = blockIdx.x * 256 + threadIdx.x;      // 1024*8 = 8192
  int d = idx >> 3, h = idx & 7;
  const float* wg = Wg + (size_t)d * (H * F) + h * F;
  const float* we = We + (size_t)d * (H * F) + h * F;
  float s1 = 0.f, s2 = 0.f, s3 = 0.f;
  for (int f = 0; f < F; ++f) {
    s1 += wg[f] * w_attn[f];
    s2 += wg[f] * w_attn[F + f];
    s3 += we[f] * w_attn[2 * F + f];
  }
  wgw1[idx] = s1; wgw2[idx] = s2; wew3[idx] = s3;
}

// ---------------------------------------------------------------------------
// K_scalars: a1/a2 = obj_vecs @ wgw{1,2} ; s1/s2 = per-head dot of obj_embds
// ---------------------------------------------------------------------------
__global__ void k_scalars(const float* __restrict__ obj, const float* __restrict__ oe,
                          const float* __restrict__ wgw1, const float* __restrict__ wgw2,
                          const float* __restrict__ u_score,
                          float* __restrict__ a1, float* __restrict__ a2,
                          float* __restrict__ s1o, float* __restrict__ s2o) {
  int idx = blockIdx.x * 256 + threadIdx.x;      // B*N*H = 32768
  int r = idx >> 3, h = idx & 7;
  const float* x = obj + (size_t)r * D;
  float va1 = 0.f, va2 = 0.f;
  for (int d2 = 0; d2 < D; ++d2) {
    float xv = x[d2];
    va1 += xv * wgw1[d2 * H + h];
    va2 += xv * wgw2[d2 * H + h];
  }
  const float* y = oe + (size_t)r * D + h * F;
  float vs1 = 0.f, vs2 = 0.f;
  for (int f = 0; f < F; ++f) { vs1 += y[f] * u_score[f]; vs2 += y[f] * u_score[F + f]; }
  a1[idx] = va1; a2[idx] = va2; s1o[idx] = vs1; s2o[idx] = vs2;
}

// ---------------------------------------------------------------------------
// K_a3e: a3e[b,e,h] = new_rela[b,e,:] . wew3[:,h]
// ---------------------------------------------------------------------------
__global__ void k_a3e(const float* __restrict__ new_rela, const float* __restrict__ wew3,
                      float* __restrict__ a3e) {
  int idx = blockIdx.x * 256 + threadIdx.x;      // B*E*H = 524288
  int r = idx >> 3, h = idx & 7;
  const float* x = new_rela + (size_t)r * D;
  float s = 0.f;
  for (int d2 = 0; d2 < D; ++d2) s += x[d2] * wew3[d2 * H + h];
  a3e[idx] = s;
}

// ---------------------------------------------------------------------------
// Edge-id map: eid[b, i*N+j] = edge index or -1
// ---------------------------------------------------------------------------
__global__ void k_eid_init(int* __restrict__ eid) {
  eid[blockIdx.x * 256 + threadIdx.x] = -1;      // B*N*N = 1048576
}
__global__ void k_eid_scatter(const int* __restrict__ edges, int* __restrict__ eid) {
  int r = blockIdx.x * 256 + threadIdx.x;        // B*E = 65536
  int b = r >> 12;
  int s = edges[r * 2 + 0], o = edges[r * 2 + 1];
  eid[(size_t)b * (N * N) + s * N + o] = r & (E - 1);
}

// ---------------------------------------------------------------------------
// K_softmax: one wave per (b,i,h) row of N=256; dense softmax matching the
// reference (-1000 filler where no edge), emits fp16 attention weights.
// ---------------------------------------------------------------------------
__global__ void k_softmax(const float* __restrict__ a1, const float* __restrict__ a2,
                          const float* __restrict__ s1, const float* __restrict__ s2,
                          const float* __restrict__ a3e, const int* __restrict__ eid,
                          _Float16* __restrict__ att16) {
  int bi = blockIdx.x;                           // b*N + i, 4096 blocks
  int b = bi >> 8, i = bi & 255;
  int h = threadIdx.x >> 5;
  int lane = threadIdx.x & 31;
  float a1v = a1[bi * H + h];
  float s1v = s1[bi * H + h];
  const int* em = eid + (size_t)b * (N * N) + (size_t)i * N;
  float vals[8];
  float m = -3.0e38f;
  #pragma unroll
  for (int it = 0; it < 8; ++it) {
    int j = lane + it * 32;
    int idx = em[j];
    float e;
    if (idx >= 0) {
      float t = a1v + a2[(b * N + j) * H + h] + a3e[((size_t)b * E + idx) * H + h];
      t = leaky(t);
      float w = fmaxf(s1v + s2[(b * N + j) * H + h], 0.f);
      e = t * w;
    } else {
      e = -1000.f;
    }
    vals[it] = e;
    m = fmaxf(m, e);
  }
  #pragma unroll
  for (int off = 16; off > 0; off >>= 1) m = fmaxf(m, __shfl_xor(m, off, 32));
  float sum = 0.f;
  #pragma unroll
  for (int it = 0; it < 8; ++it) { vals[it] = expf(vals[it] - m); sum += vals[it]; }
  #pragma unroll
  for (int off = 16; off > 0; off >>= 1) sum += __shfl_xor(sum, off, 32);
  float inv = 1.0f / sum;
  _Float16* orow = att16 + ((size_t)(b * H + h)) * (N * N) + (size_t)i * N;
  #pragma unroll
  for (int it = 0; it < 8; ++it) orow[lane + it * 32] = (_Float16)(vals[it] * inv);
}

// ---------------------------------------------------------------------------
// Templated WMMA GEMM (f16 inputs in LDS, f32 accumulate).
// Block = 256 threads = 8 waves; block tile 32(M) x 128(N); K stepped by 64.
// Wave (2x4 layout) owns a 16x32 C strip: 2 accumulators, 4 WMMAs per barrier.
// A staged row-major [row][k] (pad 72 -> all 8-half chunks 16B aligned);
// B staged TRANSPOSED [col][k] so fragments load as 2x ds_load_b128.
//
// MODE 1: new_attr = relu([obj|attr] @ W_attr + b) + attr          (K=2048)
// MODE 2: new_rela = (relu([subj|rela|obj] @ W_rela + b) + rela)*m (K=3072, gathered)
// MODE 3: g(f16)   = obj @ Wg                                      (K=1024)
// MODE 4: new_obj  = leaky(att @ g_h) batched over (b,h)           (K=256)
// ---------------------------------------------------------------------------
template<int MODE>
__global__ __launch_bounds__(256)
void gemm_wmma(const float* __restrict__ A0, const float* __restrict__ A1,
               const float* __restrict__ A2, const int* __restrict__ edges,
               const float* __restrict__ W, const float* __restrict__ bias,
               const float* __restrict__ resid, const float* __restrict__ maskp,
               const _Float16* __restrict__ att16, const _Float16* __restrict__ g16,
               float* __restrict__ outF, _Float16* __restrict__ outH,
               int K, int ldw) {
  __shared__ __align__(16) _Float16 lds_a[32][72];    // 32 rows x 64 K (pad 72)
  __shared__ __align__(16) _Float16 lds_bt[128][72];  // 128 cols x 64 K (pad 72)

  const int tid  = threadIdx.x;
  const int lane = tid & 31, wid = tid >> 5;
  const int wm = (wid >> 2) * 16;        // 0 or 16
  const int wn = (wid & 3) * 32;         // 0,32,64,96
  const size_t rowBase = (size_t)blockIdx.y * 32;
  const int colBase = blockIdx.x * 128;

  int bb = 0, hh = 0;
  size_t bh = 0;
  if constexpr (MODE == 4) { bh = blockIdx.z; bb = (int)(bh >> 3); hh = (int)(bh & 7); }

  v8f acc0 = {}, acc1 = {};
  for (int k0 = 0; k0 < K; k0 += 64) {
    // ---- stage A tile: 32 rows x 64 K as f16 pairs (pairs never straddle
    //      the 1024-aligned gather segments since pair starts are even) ----
    #pragma unroll
    for (int i = 0; i < 4; ++i) {
      int p = tid + i * 256;             // 1024 pairs
      int rl = p >> 5;
      int kp = (p & 31) * 2;
      size_t gr = rowBase + rl;
      int gk = k0 + kp;
      if constexpr (MODE == 4) {
        *(hh2*)&lds_a[rl][kp] = *(const hh2*)&att16[bh * (N * N) + gr * N + gk];
      } else {
        ff2 v;
        if constexpr (MODE == 1) {
          v = (gk < D) ? *(const ff2*)&A0[gr * D + gk]
                       : *(const ff2*)&A1[gr * D + (gk - D)];
        } else if constexpr (MODE == 2) {
          int b2 = (int)(gr >> 12);
          if (gk < D) {
            int s = edges[gr * 2 + 0];
            v = *(const ff2*)&A0[((size_t)(b2 * N + s)) * D + gk];
          } else if (gk < 2 * D) {
            v = *(const ff2*)&A2[gr * D + (gk - D)];
          } else {
            int o = edges[gr * 2 + 1];
            v = *(const ff2*)&A0[((size_t)(b2 * N + o)) * D + (gk - 2 * D)];
          }
        } else {
          v = *(const ff2*)&A0[gr * D + gk];
        }
        hh2 hv; hv.x = (_Float16)v.x; hv.y = (_Float16)v.y;
        *(hh2*)&lds_a[rl][kp] = hv;
      }
    }
    // ---- stage B tile transposed: [col][k], col-fast global mapping ----
    #pragma unroll 4
    for (int i = 0; i < 16; ++i) {
      int p = tid + i * 256;             // 4096 pairs: 128 cols x 32 k-pairs
      int kr = (p >> 7) * 2;
      int cl = p & 127;
      hh2 hv;
      if constexpr (MODE == 4) {
        size_t base = ((size_t)(bb * N + k0 + kr)) * (H * F) + hh * F + colBase + cl;
        hv.x = g16[base];
        hv.y = g16[base + (H * F)];
      } else {
        size_t base = (size_t)(k0 + kr) * ldw + colBase + cl;
        hv.x = (_Float16)W[base];
        hv.y = (_Float16)W[base + ldw];
      }
      *(hh2*)&lds_bt[cl][kr] = hv;
    }
    __syncthreads();

    // ---- 2 K-substeps x 2 N-tiles = 4 WMMAs per barrier pair ----
    const int row  = wm + (lane & 15);
    const int colB = wn + (lane & 15);
    #pragma unroll
    for (int s = 0; s < 2; ++s) {
      // A fragment: lanes 0-15 hold K 0..15; lanes 16-31 K 16..31
      int ka = s * 32 + ((lane & 16) ? 16 : 0);
      v8h a_lo = *(const v8h*)&lds_a[row][ka];
      v8h a_hi = *(const v8h*)&lds_a[row][ka + 8];
      v16h af = __builtin_shufflevector(a_lo, a_hi, 0, 1, 2, 3, 4, 5, 6, 7,
                                        8, 9, 10, 11, 12, 13, 14, 15);
      // B fragments: lanes 0-15 hold K 0..15 of col; lanes 16-31 K 16..31
      int kb = s * 32 + ((lane & 16) ? 16 : 0);
      v8h b0l = *(const v8h*)&lds_bt[colB][kb];
      v8h b0h = *(const v8h*)&lds_bt[colB][kb + 8];
      v16h bf0 = __builtin_shufflevector(b0l, b0h, 0, 1, 2, 3, 4, 5, 6, 7,
                                         8, 9, 10, 11, 12, 13, 14, 15);
      v8h b1l = *(const v8h*)&lds_bt[colB + 16][kb];
      v8h b1h = *(const v8h*)&lds_bt[colB + 16][kb + 8];
      v16h bf1 = __builtin_shufflevector(b1l, b1h, 0, 1, 2, 3, 4, 5, 6, 7,
                                         8, 9, 10, 11, 12, 13, 14, 15);
      acc0 = __builtin_amdgcn_wmma_f32_16x16x32_f16(false, af, false, bf0,
                                                    (short)0, acc0, false, false);
      acc1 = __builtin_amdgcn_wmma_f32_16x16x32_f16(false, af, false, bf1,
                                                    (short)0, acc1, false, false);
    }
    __syncthreads();
  }

  // ---- epilogue (C layout: vgpr r -> row r (lanes<16) / r+8 (lanes>=16)) ----
  const int cl0 = wn + (lane & 15);
  const int rb = wm + ((lane & 16) ? 8 : 0);
  #pragma unroll
  for (int t = 0; t < 2; ++t) {
    const v8f& acc = t ? acc1 : acc0;
    const int gc = colBase + cl0 + t * 16;
    #pragma unroll
    for (int vr = 0; vr < 8; ++vr) {
      size_t gr = rowBase + rb + vr;
      float v = acc[vr];
      if constexpr (MODE == 1) {
        v = fmaxf(v + bias[gc], 0.f) + resid[gr * D + gc];
        outF[gr * D + gc] = v;
      } else if constexpr (MODE == 2) {
        v = (fmaxf(v + bias[gc], 0.f) + resid[gr * D + gc]) * maskp[gr];
        outF[gr * D + gc] = v;
      } else if constexpr (MODE == 3) {
        outH[gr * (H * F) + gc] = (_Float16)v;
      } else {
        outF[((size_t)bb * N + gr) * (H * F) + hh * F + gc] = leaky(v);
      }
    }
  }
}

// ---------------------------------------------------------------------------
extern "C" void kernel_launch(void* const* d_in, const int* in_sizes, int n_in,
                              void* d_out, int out_size, void* d_ws, size_t ws_size,
                              hipStream_t stream) {
  const float* obj_vecs   = (const float*)d_in[0];
  const float* attr_vecs  = (const float*)d_in[1];
  const float* rela_vecs  = (const float*)d_in[2];
  const float* rela_masks = (const float*)d_in[3];
  const float* obj_embds  = (const float*)d_in[4];
  const float* W_attr     = (const float*)d_in[5];
  const float* b_attr     = (const float*)d_in[6];
  const float* W_rela     = (const float*)d_in[7];
  const float* b_rela     = (const float*)d_in[8];
  const float* Wg         = (const float*)d_in[9];
  const float* We         = (const float*)d_in[10];
  const float* w_attn     = (const float*)d_in[11];
  const float* u_score    = (const float*)d_in[12];
  const int*   edges      = (const int*)d_in[13];

  float* out_obj  = (float*)d_out;                       // (B,N,H*F)
  float* out_attr = out_obj + (size_t)B * N * D;         // (B,N,D)
  float* out_rela = out_attr + (size_t)B * N * D;        // (B,E,D)

  // workspace layout (~31 MB)
  char* ws = (char*)d_ws;
  size_t off = 0;
  auto alloc = [&](size_t bytes) { size_t o = off; off += (bytes + 255) & ~(size_t)255; return o; };
  float*    wgw1  = (float*)(ws + alloc((size_t)D * H * 4));
  float*    wgw2  = (float*)(ws + alloc((size_t)D * H * 4));
  float*    wew3  = (float*)(ws + alloc((size_t)D * H * 4));
  float*    a1    = (float*)(ws + alloc((size_t)B * N * H * 4));
  float*    a2    = (float*)(ws + alloc((size_t)B * N * H * 4));
  float*    s1    = (float*)(ws + alloc((size_t)B * N * H * 4));
  float*    s2    = (float*)(ws + alloc((size_t)B * N * H * 4));
  float*    a3e   = (float*)(ws + alloc((size_t)B * E * H * 4));
  int*      eid   = (int*)  (ws + alloc((size_t)B * N * N * 4));
  _Float16* g16   = (_Float16*)(ws + alloc((size_t)B * N * H * F * 2));
  _Float16* att16 = (_Float16*)(ws + alloc((size_t)B * H * N * N * 2));

  // 1) fold attention vectors into projection matrices
  k_proj<<<32, 256, 0, stream>>>(Wg, We, w_attn, wgw1, wgw2, wew3);
  // 2) per-node attention scalars (a1,a2) and score scalars (s1,s2)
  k_scalars<<<128, 256, 0, stream>>>(obj_vecs, obj_embds, wgw1, wgw2, u_score,
                                     a1, a2, s1, s2);
  // 3) new_attr = relu([obj|attr] @ W_attr + b) + attr         (WMMA)
  gemm_wmma<1><<<dim3(D / 128, (B * N) / 32, 1), 256, 0, stream>>>(
      obj_vecs, attr_vecs, nullptr, nullptr, W_attr, b_attr, attr_vecs, nullptr,
      nullptr, nullptr, out_attr, nullptr, 2 * D, D);
  // 4) new_rela = (relu(triplet @ W_rela + b) + rela) * mask   (WMMA, fused gather)
  gemm_wmma<2><<<dim3(D / 128, (B * E) / 32, 1), 256, 0, stream>>>(
      obj_vecs, nullptr, rela_vecs, edges, W_rela, b_rela, rela_vecs, rela_masks,
      nullptr, nullptr, out_rela, nullptr, 3 * D, D);
  // 5) a3e = new_rela @ (We . w3)
  k_a3e<<<(B * E * H) / 256, 256, 0, stream>>>(out_rela, wew3, a3e);
  // 6) dense edge-id map
  k_eid_init<<<(B * N * N) / 256, 256, 0, stream>>>(eid);
  k_eid_scatter<<<(B * E) / 256, 256, 0, stream>>>(edges, eid);
  // 7) g = obj_vecs @ Wg  (fp16 output for attention apply)    (WMMA)
  gemm_wmma<3><<<dim3((H * F) / 128, (B * N) / 32, 1), 256, 0, stream>>>(
      obj_vecs, nullptr, nullptr, nullptr, Wg, nullptr, nullptr, nullptr,
      nullptr, nullptr, nullptr, g16, D, H * F);
  // 8) masked softmax -> fp16 attention weights
  k_softmax<<<B * N, 256, 0, stream>>>(a1, a2, s1, s2, a3e, eid, att16);
  // 9) new_obj = leaky(att @ g_h), batched over (b,h)          (WMMA)
  gemm_wmma<4><<<dim3(F / 128, N / 32, B * H), 256, 0, stream>>>(
      nullptr, nullptr, nullptr, nullptr, nullptr, nullptr, nullptr, nullptr,
      att16, g16, out_obj, nullptr, N, 0);

  (void)in_sizes; (void)n_in; (void)out_size; (void)ws_size;
}
